// VirtualNode_39024072851536
// MI455X (gfx1250) — matrix operand order repaired and checked by reading, not compiled, not run on hardware
//
#include <hip/hip_runtime.h>
#include <hip/hip_bf16.h>
#include <math.h>

// ---------------- types ----------------
typedef __bf16 bf16_t;
typedef __attribute__((ext_vector_type(16))) __bf16 v16bf;
typedef __attribute__((ext_vector_type(8)))  __bf16 v8bf;
typedef __attribute__((ext_vector_type(8)))  float  v8f;
typedef __attribute__((ext_vector_type(4)))  float  v4f;

// ---------------- problem constants ----------------
#define N_NODES 524288
#define NV 64          // clusters (virtual nodes)
#define NC 8192        // nodes per cluster
#define DD 256         // feature dim
#define HH 256         // hidden

// ---------------- workspace layout (bytes) ----------------
#define WS_Q      0u          // 64*256 f32      (65536 B)
#define WS_WKT    65536u      // 256*256 bf16    (131072 B)
#define WS_WTS    196608u     // N f32 scores/weights (2 MB)
#define WS_PART   2293760u    // 8*64*256 f32 partial pools (524288 B)
#define WS_Z      2818048u    // 64*512 f32 MLP hidden (131072 B)
// total ~2.95 MB

// ---------------- fast branchless transcendentals ----------------
// tanh(x) = 1 - 2/(exp2(2*log2(e)*x) + 1); v_exp_f32 saturates cleanly
// at both ends (inf -> 1, 0 -> -1), so no range branches needed.
__device__ __forceinline__ float fast_tanh(float x) {
    const float e = __builtin_amdgcn_exp2f(x * 2.88539008177793f);
    return 1.f - 2.f * __builtin_amdgcn_rcpf(e + 1.f);
}
// exp(x) for x <= 0 (softmax-safe): single v_exp_f32
__device__ __forceinline__ float fast_exp_nonpos(float x) {
    return __builtin_amdgcn_exp2f(x * 1.44269504088896f);
}

// ============================================================
// kernel 0a: q[v][e] = sum_d vn_emb[v][d] * wq[d][e]
// ============================================================
__global__ void __launch_bounds__(256) k_q(const float* __restrict__ vx,
                                           const float* __restrict__ wq,
                                           float* __restrict__ q) {
    const int v = blockIdx.x, e = threadIdx.x;
    const float* row = vx + v * DD;
    float acc = 0.f;
    for (int d = 0; d < DD; ++d) acc += row[d] * wq[d * DD + e];
    q[v * DD + e] = acc;
}

// ============================================================
// kernel 0b: wkT[e][d] = bf16(wk[d][e])   (N-major B operand)
// ============================================================
__global__ void __launch_bounds__(256) k_wkT(const float* __restrict__ wk,
                                             bf16_t* __restrict__ wkT) {
    const int e = blockIdx.x, d = threadIdx.x;
    wkT[e * DD + d] = (bf16_t)wk[d * DD + e];
}

// ============================================================
// kernel 1 (fused stream + WMMA):
//   h = x + vn_emb[v]  -> store f32, stage bf16 tile in LDS
//   kproj = h @ wk via v_wmma_f32_16x16x32_bf16
//   scores[node] = sum_e tanh(kproj[e] + q[v][e]) * v_attn[e]
// 128 rows per 256-thread block (8 waves x 16-row tiles).
// ============================================================
__global__ void __launch_bounds__(256) k_fuse(
    const float*  __restrict__ x,
    const float*  __restrict__ vn_emb,
    const float*  __restrict__ q,
    const float*  __restrict__ v_attn,
    const bf16_t* __restrict__ wkT,
    float* __restrict__ h_out,
    float* __restrict__ scores)
{
    __shared__ bf16_t At[128 * DD];          // 64 KB bf16 h-tile

    const int t    = threadIdx.x;
    const int row0 = blockIdx.x * 128;
    const int v    = row0 >> 13;             // row0 / 8192
    const float* vx = vn_emb + v * DD;

    // ---- streaming phase: h = x + vx ; store + stage ----
    for (int i = 0; i < 32; ++i) {
        const int idx = t + (i << 8);        // float4 index in 128x64 grid
        const int row = idx >> 6;
        const int c4  = idx & 63;
        const v4f* xp = (const v4f*)(x + (size_t)(row0 + row) * DD) + c4;
        // prefetch the x stream a few KB ahead (global_prefetch_b8)
        __builtin_prefetch((const char*)xp + 16384, 0, 1);
        v4f xv = *xp;
        v4f vv = ((const v4f*)vx)[c4];
        v4f hv = xv + vv;
        ((v4f*)(h_out + (size_t)(row0 + row) * DD))[c4] = hv;
        const int cb = c4 << 2;
        At[row * DD + cb + 0] = (bf16_t)hv.x;
        At[row * DD + cb + 1] = (bf16_t)hv.y;
        At[row * DD + cb + 2] = (bf16_t)hv.z;
        At[row * DD + cb + 3] = (bf16_t)hv.w;
    }
    __syncthreads();

    // ---- WMMA phase ----
    const int lane = t & 31;
    const int wave = t >> 5;
    const int rw   = wave << 4;              // 16-row tile base for this wave
    const int m    = lane & 15;
    const int hi   = lane >> 4;

    float sacc[8];
#pragma unroll
    for (int r = 0; r < 8; ++r) sacc[r] = 0.f;

    const float* qv = q + v * DD;

    for (int nt = 0; nt < 16; ++nt) {
        v8f c = {};
        const int n = (nt << 4) + m;
#pragma unroll
        for (int ks = 0; ks < 8; ++ks) {
            const int k0 = ks << 5;
            // A fragment (16x32 bf16): lane<16 -> K {k0..k0+15}
            //                          lane>=16 -> K {k0+16..k0+31}
            const bf16_t* ap = &At[(rw + m) * DD + k0 + hi * 16];
            v8bf a0 = *(const v8bf*)(ap);
            v8bf a1 = *(const v8bf*)(ap + 8);
            v16bf a;
#pragma unroll
            for (int i = 0; i < 8; ++i) { a[i] = a0[i]; a[i + 8] = a1[i]; }
            // B fragment (32x16 bf16, N-major wkT): 16 contiguous K per lane
            v16bf b = *(const v16bf*)(wkT + n * DD + k0 + hi * 16);
            c = __builtin_amdgcn_wmma_f32_16x16x32_bf16(
                    false, a, false, b, (short)0, c, false, false);
        }
        // epilogue on C tile: lane holds N = n, M = r + 8*hi (branchless)
        const float qn = qv[n];
        const float va = v_attn[n];
#pragma unroll
        for (int r = 0; r < 8; ++r) sacc[r] += fast_tanh(c[r] + qn) * va;
    }

    // reduce over the 16 lanes of each half (same M set per half)
#pragma unroll
    for (int mask = 1; mask < 16; mask <<= 1) {
#pragma unroll
        for (int r = 0; r < 8; ++r) sacc[r] += __shfl_xor(sacc[r], mask, 32);
    }
    if (m == 0) {                            // lanes 0 (rows rw..rw+7) and 16 (rw+8..rw+15)
        const int rbase = row0 + rw + (hi << 3);
#pragma unroll
        for (int r = 0; r < 8; ++r) scores[rbase + r] = sacc[r];
    }
}

// ============================================================
// kernel 2: per-cluster softmax, in place (scores -> weights)
// ============================================================
__global__ void __launch_bounds__(256) k_soft(float* __restrict__ s_io) {
    __shared__ float red[256];
    const int v = blockIdx.x, t = threadIdx.x;
    float* s = s_io + (size_t)v * NC;

    float mx = -3.4e38f;
    for (int i = t; i < NC; i += 256) mx = fmaxf(mx, s[i]);
    red[t] = mx; __syncthreads();
    for (int off = 128; off > 0; off >>= 1) {
        if (t < off) red[t] = fmaxf(red[t], red[t + off]);
        __syncthreads();
    }
    mx = red[0]; __syncthreads();

    float sum = 0.f;
    for (int i = t; i < NC; i += 256) sum += fast_exp_nonpos(s[i] - mx);
    red[t] = sum; __syncthreads();
    for (int off = 128; off > 0; off >>= 1) {
        if (t < off) red[t] += red[t + off];
        __syncthreads();
    }
    const float inv = 1.f / red[0];
    for (int i = t; i < NC; i += 256) s[i] = fast_exp_nonpos(s[i] - mx) * inv;
}

// ============================================================
// kernel 3: partial pooled sums (deterministic, no atomics)
//   part[sp][v][col] = sum over 1024-row split of w[c]*h[c][col]
// ============================================================
__global__ void __launch_bounds__(256) k_pool(const float* __restrict__ h,
                                              const float* __restrict__ wts,
                                              float* __restrict__ part) {
    const int v  = blockIdx.x >> 3;
    const int sp = blockIdx.x & 7;
    const int t  = threadIdx.x;
    const float* hv = h   + (size_t)v * NC * DD;
    const float* wv = wts + (size_t)v * NC;
    float acc = 0.f;
    const int r0 = sp * (NC / 8);
    for (int r = r0; r < r0 + (NC / 8); ++r) {
        __builtin_prefetch((const char*)(hv + (size_t)r * DD + t) + 8192, 0, 1);
        acc += wv[r] * hv[(size_t)r * DD + t];
    }
    part[(sp * NV + v) * DD + t] = acc;
}

// ============================================================
// kernel 4: pooled = sum(part)+vn_emb; MLP + BN + ReLU x2
// single block, register-blocked over the 64 virtual nodes
// ============================================================
__global__ void __launch_bounds__(512) k_mlp(
    const float* __restrict__ part, const float* __restrict__ vn_emb,
    const float* __restrict__ w1, const float* __restrict__ b1,
    const float* __restrict__ g1, const float* __restrict__ be1,
    const float* __restrict__ w2, const float* __restrict__ b2,
    const float* __restrict__ g2, const float* __restrict__ be2,
    float* __restrict__ Zg, float* __restrict__ out)
{
    __shared__ float P[NV * DD];             // 64 KB pooled
    const int t = threadIdx.x;
    for (int i = t; i < NV * DD; i += 512) {
        float a = vn_emb[i];
#pragma unroll
        for (int sp = 0; sp < 8; ++sp) a += part[sp * NV * DD + i];
        P[i] = a;
    }
    __syncthreads();

    // ---- layer 1: z = P @ w1 + b1 ; BN over V ; ReLU -> Zg ----
    {
        const int j = t;                     // 512 output columns
        float zs[NV];
#pragma unroll
        for (int v = 0; v < NV; ++v) zs[v] = b1[j];
        for (int k = 0; k < DD; ++k) {
            const float wv = w1[k * (2 * HH) + j];
#pragma unroll
            for (int v = 0; v < NV; ++v) zs[v] += P[v * DD + k] * wv;
        }
        float mean = 0.f, m2 = 0.f;
#pragma unroll
        for (int v = 0; v < NV; ++v) { mean += zs[v]; m2 += zs[v] * zs[v]; }
        mean *= (1.f / NV);
        const float var = m2 * (1.f / NV) - mean * mean;
        const float sc = rsqrtf(var + 1e-5f) * g1[j];
        const float sh = be1[j] - mean * sc;
#pragma unroll
        for (int v = 0; v < NV; ++v)
            Zg[v * (2 * HH) + j] = fmaxf(zs[v] * sc + sh, 0.f);
    }
    __syncthreads();

    // ---- layer 2: y = Z @ w2 + b2 ; BN over V ; ReLU -> out ----
    if (t < HH) {
        const int j = t;
        float ys[NV];
#pragma unroll
        for (int v = 0; v < NV; ++v) ys[v] = b2[j];
        for (int k = 0; k < 2 * HH; ++k) {
            const float wv = w2[k * HH + j];
#pragma unroll
            for (int v = 0; v < NV; ++v) ys[v] += Zg[v * (2 * HH) + k] * wv;
        }
        float mean = 0.f, m2 = 0.f;
#pragma unroll
        for (int v = 0; v < NV; ++v) { mean += ys[v]; m2 += ys[v] * ys[v]; }
        mean *= (1.f / NV);
        const float var = m2 * (1.f / NV) - mean * mean;
        const float sc = rsqrtf(var + 1e-5f) * g2[j];
        const float sh = be2[j] - mean * sc;
#pragma unroll
        for (int v = 0; v < NV; ++v)
            out[v * HH + j] = fmaxf(ys[v] * sc + sh, 0.f);
    }
}

// ============================================================
// launch
// ============================================================
extern "C" void kernel_launch(void* const* d_in, const int* in_sizes, int n_in,
                              void* d_out, int out_size, void* d_ws, size_t ws_size,
                              hipStream_t stream) {
    const float* x      = (const float*)d_in[0];
    const float* vn_emb = (const float*)d_in[1];
    const float* wq     = (const float*)d_in[2];
    const float* wk     = (const float*)d_in[3];
    const float* v_attn = (const float*)d_in[4];
    const float* w1     = (const float*)d_in[5];
    const float* b1     = (const float*)d_in[6];
    const float* g1     = (const float*)d_in[7];
    const float* be1    = (const float*)d_in[8];
    const float* w2     = (const float*)d_in[9];
    const float* b2     = (const float*)d_in[10];
    const float* g2     = (const float*)d_in[11];
    const float* be2    = (const float*)d_in[12];
    // d_in[13] vn_index / d_in[14] vn_indices: identity block mapping (v = node / 8192)

    char* ws = (char*)d_ws;
    float*  q    = (float*)(ws + WS_Q);
    bf16_t* wkT  = (bf16_t*)(ws + WS_WKT);
    float*  wts  = (float*)(ws + WS_WTS);
    float*  part = (float*)(ws + WS_PART);
    float*  Zg   = (float*)(ws + WS_Z);

    float* h_out   = (float*)d_out;
    float* vxn_out = h_out + (size_t)N_NODES * DD;

    hipLaunchKernelGGL(k_q,    dim3(NV),            dim3(DD),  0, stream, vn_emb, wq, q);
    hipLaunchKernelGGL(k_wkT,  dim3(DD),            dim3(DD),  0, stream, wk, wkT);
    hipLaunchKernelGGL(k_fuse, dim3(N_NODES / 128), dim3(256), 0, stream,
                       x, vn_emb, q, v_attn, wkT, h_out, wts);
    hipLaunchKernelGGL(k_soft, dim3(NV),            dim3(256), 0, stream, wts);
    hipLaunchKernelGGL(k_pool, dim3(NV * 8),        dim3(DD),  0, stream, h_out, wts, part);
    hipLaunchKernelGGL(k_mlp,  dim3(1),             dim3(512), 0, stream,
                       part, vn_emb, w1, b1, g1, be1, w2, b2, g2, be2, Zg, vxn_out);
}
